// GNN_40785009443055
// MI455X (gfx1250) — matrix operand order repaired and checked by reading, not compiled, not run on hardware
//
#include <hip/hip_runtime.h>
#include <hip/hip_bf16.h>

typedef _Float16 half_t;
typedef __attribute__((ext_vector_type(16))) _Float16 v16h;
typedef __attribute__((ext_vector_type(8)))  _Float16 v8h;
typedef __attribute__((ext_vector_type(8)))  float    v8f;

#define NN     265
#define PASTN  200
#define FUTN   65
#define BB     1024
#define DEG_EPS 1e-15f
#define BN_EPS  1e-5f

// ---------------------------------------------------------------------------
// WMMA helpers (CDNA5 v_wmma_f32_16x16x32_f16, wave32)
// A 16x32 f16: lane L<16 -> row L, K 0..15 ; lane L+16 -> K 16..31
// B 32x16 f16: lane = column N (&15); lanes<16 hold K 0..15, lanes>=16 hold K 16..31
// C/D f32: VGPR q: lanes<16 -> M=q, N=lane ; lanes>=16 -> M=q+8, N=lane-16
// ---------------------------------------------------------------------------
__device__ __forceinline__ v16h frag_a(const half_t* base, int ld, int lane, bool rowValid) {
  v16h a;
  if (rowValid) {
    const half_t* p = base + (long)(lane & 15) * ld + ((lane < 16) ? 0 : 16);
#pragma unroll
    for (int i = 0; i < 16; ++i) a[i] = p[i];
  } else {
#pragma unroll
    for (int i = 0; i < 16; ++i) a[i] = (half_t)0.f;
  }
  return a;
}

__device__ __forceinline__ v16h frag_b(const half_t* W, int ld, int n0, int k0, int lane) {
  const half_t* p = W + (long)(n0 + (lane & 15)) * ld + k0 + ((lane < 16) ? 0 : 16);
  v16h b;
#pragma unroll
  for (int i = 0; i < 16; ++i) b[i] = p[i];
  return b;
}

__device__ __forceinline__ v8f wmma16(v16h a, v16h b, v8f c) {
  return __builtin_amdgcn_wmma_f32_16x16x32_f16(false, a, false, b, (short)0, c, false, false);
}

// ---------------------------------------------------------------------------
// Weight convert + zero-pad to half:  dst[rowsPad][ldk]
// ---------------------------------------------------------------------------
__global__ void cvt_pad_kernel(const float* __restrict__ src, half_t* __restrict__ dst,
                               int rows, int cols, int ldk, int rowsPad) {
  int i = blockIdx.x * 256 + threadIdx.x;
  int total = rowsPad * ldk;
  if (i >= total) return;
  int r = i / ldk, c = i % ldk;
  float v = (r < rows && c < cols) ? src[r * cols + c] : 0.f;
  dst[i] = (half_t)v;
}

__global__ void zero_half_kernel(half_t* __restrict__ p, long n) {
  long i = (long)blockIdx.x * 256 + threadIdx.x;
  if (i < n) p[i] = (half_t)0.f;
}

// ---------------------------------------------------------------------------
// Adjacency: S = relu(M M^T), masked (j<i) softmax per row -> A; plus full degrees
// ---------------------------------------------------------------------------
__global__ void adj_kernel(const float* __restrict__ M, float* __restrict__ A,
                           float* __restrict__ dfull) {
  const int i = blockIdx.x, t = threadIdx.x;
  __shared__ float mrow[NN];
  __shared__ float srow[NN];
  __shared__ float red[256];
  for (int k = t; k < NN; k += 256) mrow[k] = M[(long)i * NN + k];
  __syncthreads();
  for (int j = t; j < NN; j += 256) {
    float s = 0.f;
    const float* mj = M + (long)j * NN;
    for (int k = 0; k < NN; ++k) s += mrow[k] * mj[k];
    srow[j] = fmaxf(s, 0.f);
  }
  __syncthreads();
  float mx = -3.0e38f;
  for (int j = t; j < i; j += 256) mx = fmaxf(mx, srow[j]);
  red[t] = mx; __syncthreads();
  for (int o = 128; o > 0; o >>= 1) { if (t < o) red[t] = fmaxf(red[t], red[t + o]); __syncthreads(); }
  mx = red[0]; __syncthreads();
  float sum = 0.f;
  for (int j = t; j < i; j += 256) sum += __expf(srow[j] - mx);
  red[t] = sum; __syncthreads();
  for (int o = 128; o > 0; o >>= 1) { if (t < o) red[t] += red[t + o]; __syncthreads(); }
  sum = red[0]; __syncthreads();
  float inv = (i > 0 && sum > 0.f) ? 1.0f / sum : 0.f;
  int cnt = 0;
  for (int j = t; j < NN; j += 256) {
    float a = (j < i) ? __expf(srow[j] - mx) * inv : 0.f;
    A[(long)i * NN + j] = a;
    if (a > DEG_EPS) cnt++;
  }
  red[t] = (float)cnt; __syncthreads();
  for (int o = 128; o > 0; o >>= 1) { if (t < o) red[t] += red[t + o]; __syncthreads(); }
  if (t == 0) dfull[i] = rsqrtf(1.0f + red[0]);
}

// Sub-graph degrees: d11 (A11 rows), d22 (A22 rows), dp (A[200:,:] rows)
__global__ void subdeg_kernel(const float* __restrict__ A, float* __restrict__ d11,
                              float* __restrict__ d22, float* __restrict__ dp) {
  __shared__ float red[256];
  int bid = blockIdx.x, t = threadIdx.x;
  int cnt = 0;
  if (bid < PASTN) {
    const float* row = A + (long)bid * NN;
    for (int j = t; j < PASTN; j += 256) cnt += (row[j] > DEG_EPS) ? 1 : 0;
  } else if (bid < PASTN + FUTN) {
    const float* row = A + (long)bid * NN;
    for (int j = t; j < FUTN; j += 256) cnt += (row[PASTN + j] > DEG_EPS) ? 1 : 0;
  } else {
    int m = bid - (PASTN + FUTN);
    const float* row = A + (long)(PASTN + m) * NN;
    for (int j = t; j < NN; j += 256) cnt += (row[j] > DEG_EPS) ? 1 : 0;
  }
  red[t] = (float)cnt; __syncthreads();
  for (int o = 128; o > 0; o >>= 1) { if (t < o) red[t] += red[t + o]; __syncthreads(); }
  if (t == 0) {
    float v = rsqrtf(1.0f + red[0]);
    if (bid < PASTN) d11[bid] = v;
    else if (bid < PASTN + FUTN) d22[bid - PASTN] = v;
    else dp[bid - PASTN - FUTN] = v;
  }
}

// Build padded half S matrices: S11h[224][224], S22h[96][96], Sph[96][288]
__global__ void build_s_kernel(const float* __restrict__ A, const float* __restrict__ d11,
                               const float* __restrict__ d22, const float* __restrict__ dp,
                               half_t* __restrict__ S11, half_t* __restrict__ S22,
                               half_t* __restrict__ Sp) {
  const int n11 = 224 * 224, n22 = 96 * 96, np = 96 * 288;
  int idx = blockIdx.x * 256 + threadIdx.x;
  if (idx < n11) {
    int m = idx / 224, n = idx % 224;
    float v = 0.f;
    if (m < PASTN && n < PASTN)
      v = d11[m] * (A[(long)m * NN + n] + (m == n ? 1.f : 0.f)) * d11[n];
    S11[idx] = (half_t)v;
  } else if (idx < n11 + n22) {
    int r = idx - n11;
    int m = r / 96, n = r % 96;
    float v = 0.f;
    if (m < FUTN && n < FUTN)
      v = d22[m] * (A[(long)(PASTN + m) * NN + PASTN + n] + (m == n ? 1.f : 0.f)) * d22[n];
    S22[r] = (half_t)v;
  } else if (idx < n11 + n22 + np) {
    int r = idx - n11 - n22;
    int m = r / 288, n = r % 288;
    float v = 0.f;
    if (m < FUTN && n < NN)
      v = dp[m] * dp[m] * (A[(long)(PASTN + m) * NN + n] + (n == PASTN + m ? 1.f : 0.f));
    Sp[r] = (half_t)v;
  }
}

// ---------------------------------------------------------------------------
// Pre-stage BN1 statistics: per-embedding-row sums, then per-node mean/var
// ---------------------------------------------------------------------------
__global__ void embsum_kernel(const float* __restrict__ e0, const float* __restrict__ e1,
                              const float* __restrict__ e2, float* __restrict__ rs,
                              float* __restrict__ rs2) {
  int i = threadIdx.x;
  if (i >= 70) return;
  const float* p;
  if (i < 13) p = e0 + i * 16;
  else if (i < 45) p = e1 + (i - 13) * 16;
  else p = e2 + (i - 45) * 16;
  float s = 0.f, s2 = 0.f;
  for (int c = 0; c < 16; ++c) { float v = p[c]; s += v; s2 += v * v; }
  rs[i] = s; rs2[i] = s2;
}

__global__ void nodestats_kernel(const float* __restrict__ data, const float* __restrict__ rs,
                                 const float* __restrict__ rs2, const float* __restrict__ g,
                                 const float* __restrict__ bta, float* __restrict__ scale,
                                 float* __restrict__ shiftv) {
  int n = blockIdx.x, t = threadIdx.x;
  __shared__ float r1[256], r2[256];
  float s1 = 0.f, s2 = 0.f;
  for (int b = t; b < BB; b += 256) {
    const float* dr = data + ((long)b * NN + n) * 6;
    int i0 = (int)dr[0], i1 = (int)dr[1], i2 = (int)dr[2];
    s1 += rs[i0] + rs[13 + i1] + rs[45 + i2];
    s2 += rs2[i0] + rs2[13 + i1] + rs2[45 + i2];
  }
  r1[t] = s1; r2[t] = s2; __syncthreads();
  for (int o = 128; o > 0; o >>= 1) {
    if (t < o) { r1[t] += r1[t + o]; r2[t] += r2[t + o]; }
    __syncthreads();
  }
  if (t == 0) {
    float cntv = (float)(BB * 48);
    float mu = r1[0] / cntv;
    float var = r2[0] / cntv - mu * mu;
    float sc = g[n] * rsqrtf(var + BN_EPS);
    scale[n] = sc;
    shiftv[n] = bta[n] - mu * sc;
  }
}

// ---------------------------------------------------------------------------
// Fused pre-MLP (51->128->128->50) with WMMA, one wave per 16 rows.
// Writes x_h[row][64] half (col 50 = raw last numeric, cols 51..63 = 0)
// ---------------------------------------------------------------------------
__global__ void __launch_bounds__(32) pre_mlp_kernel(
    const float* __restrict__ data, const float* __restrict__ e0,
    const float* __restrict__ e1, const float* __restrict__ e2,
    const float* __restrict__ scale, const float* __restrict__ shiftv,
    const half_t* __restrict__ W1h, const float* __restrict__ b1,
    const half_t* __restrict__ W2h, const float* __restrict__ b2,
    const half_t* __restrict__ W3h, const float* __restrict__ b3,
    half_t* __restrict__ xh) {
  __shared__ half_t hin[16][64];
  __shared__ half_t h1[16][128];
  __shared__ half_t h2[16][128];
  const int lane = threadIdx.x;
  const long tile = blockIdx.x;
  const int r = lane & 15;
  const int hi = (lane < 16) ? 0 : 8;
  {
    const long row = tile * 16 + r;
    const int n = (int)(row % NN);
    const float* dr = data + row * 6;
    int i0 = (int)dr[0], i1 = (int)dr[1], i2 = (int)dr[2];
    float sc = scale[n], sh = shiftv[n];
    int c0 = (lane < 16) ? 0 : 32;
#pragma unroll
    for (int cc = 0; cc < 32; ++cc) {
      int c = c0 + cc;
      float v;
      if (c < 16) v = sc * e0[i0 * 16 + c] + sh;
      else if (c < 32) v = sc * e1[i1 * 16 + (c - 16)] + sh;
      else if (c < 48) v = sc * e2[i2 * 16 + (c - 32)] + sh;
      else if (c < 51) v = dr[3 + (c - 48)];
      else v = 0.f;
      hin[r][c] = (half_t)v;
    }
  }
  __syncthreads();
  const int colL = lane & 15;
  // layer 1: K=64 (51 valid), 128 out
#pragma unroll
  for (int ct = 0; ct < 8; ++ct) {
    int col = ct * 16 + colL;
    v8f acc;
#pragma unroll
    for (int q = 0; q < 8; ++q) acc[q] = b1[col];
#pragma unroll
    for (int ks = 0; ks < 2; ++ks) {
      v16h a  = frag_a(&hin[0][0] + ks * 32, 64, lane, true);
      v16h bb = frag_b(W1h, 64, ct * 16, ks * 32, lane);
      acc = wmma16(a, bb, acc);
    }
#pragma unroll
    for (int q = 0; q < 8; ++q) h1[q + hi][col] = (half_t)fmaxf(acc[q], 0.f);
  }
  __syncthreads();
  // layer 2: K=128, 128 out
#pragma unroll
  for (int ct = 0; ct < 8; ++ct) {
    int col = ct * 16 + colL;
    v8f acc;
#pragma unroll
    for (int q = 0; q < 8; ++q) acc[q] = b2[col];
#pragma unroll
    for (int ks = 0; ks < 4; ++ks) {
      v16h a  = frag_a(&h1[0][0] + ks * 32, 128, lane, true);
      v16h bb = frag_b(W2h, 128, ct * 16, ks * 32, lane);
      acc = wmma16(a, bb, acc);
    }
#pragma unroll
    for (int q = 0; q < 8; ++q) h2[q + hi][col] = (half_t)fmaxf(acc[q], 0.f);
  }
  __syncthreads();
  // layer 3: K=128, 50 valid out
#pragma unroll
  for (int ct = 0; ct < 4; ++ct) {
    int col = ct * 16 + colL;
    float bias = (col < 50) ? b3[col] : 0.f;
    v8f acc;
#pragma unroll
    for (int q = 0; q < 8; ++q) acc[q] = bias;
#pragma unroll
    for (int ks = 0; ks < 4; ++ks) {
      v16h a  = frag_a(&h2[0][0] + ks * 32, 128, lane, true);
      v16h bb = frag_b(W3h, 128, ct * 16, ks * 32, lane);
      acc = wmma16(a, bb, acc);
    }
    if (col < 50) {
#pragma unroll
      for (int q = 0; q < 8; ++q) {
        long row = tile * 16 + q + hi;
        xh[row * 64 + col] = (half_t)acc[q];
      }
    }
  }
  if (lane < 16) {
    long row = tile * 16 + lane;
    xh[row * 64 + 50] = (half_t)data[row * 6 + 5];
#pragma unroll
    for (int c = 51; c < 64; ++c) xh[row * 64 + c] = (half_t)0.f;
  }
}

// ---------------------------------------------------------------------------
// Feature GEMM: out[b][f][node] (transposed, half) = X[b] @ W.T  (no bias/act)
// ---------------------------------------------------------------------------
template <int FT>
__global__ void __launch_bounds__(32) xl_gemm_kernel(
    const half_t* __restrict__ X, long xbstride, int ldx, int node_off,
    int nvalid, int npad, const half_t* __restrict__ W, int ldk,
    half_t* __restrict__ out, long obstride) {
  const int ntiles = npad >> 4;
  const int b  = blockIdx.x / ntiles;
  const int nt = blockIdx.x % ntiles;
  const int lane = threadIdx.x;
  const int ksteps = ldk >> 5;
  v8f acc[FT];
#pragma unroll
  for (int ct = 0; ct < FT; ++ct)
#pragma unroll
    for (int q = 0; q < 8; ++q) acc[ct][q] = 0.f;
  const bool rowValid = (nt * 16 + (lane & 15)) < nvalid;
  const half_t* xb = X + (long)b * xbstride + (long)(node_off + nt * 16) * ldx;
  for (int ks = 0; ks < ksteps; ++ks) {
    v16h a = frag_a(xb + ks * 32, ldx, lane, rowValid);
#pragma unroll
    for (int ct = 0; ct < FT; ++ct) {
      v16h bb = frag_b(W, ldk, ct * 16, ks * 32, lane);
      acc[ct] = wmma16(a, bb, acc[ct]);
    }
  }
  const int nbase = nt * 16 + ((lane < 16) ? 0 : 8);
#pragma unroll
  for (int ct = 0; ct < FT; ++ct) {
    const int f = ct * 16 + (lane & 15);
    v8h pk;
#pragma unroll
    for (int q = 0; q < 8; ++q) pk[q] = (half_t)acc[ct][q];
    *reinterpret_cast<v8h*>(out + (long)b * obstride + (long)f * npad + nbase) = pk;
  }
}

// ---------------------------------------------------------------------------
// Aggregation GEMM: out[b][m][f] = sum_k S[m][k] * XLT[b][f][k]
// ---------------------------------------------------------------------------
__global__ void __launch_bounds__(32) agg_gemm_kernel(
    const half_t* __restrict__ S, int Kpad, int Mpad,
    const half_t* __restrict__ XLT, long xbstride, int F, int doRelu,
    half_t* __restrict__ outH, float* __restrict__ outF32, int outMode,
    int ld_out, long obstride, int row_off, int mvalid) {
  const int mtiles = Mpad >> 4, ftiles = F >> 4;
  const int per_b = mtiles * ftiles;
  const int b = blockIdx.x / per_b;
  const int rem = blockIdx.x % per_b;
  const int mt = rem / ftiles, ft = rem % ftiles;
  const int lane = threadIdx.x;
  v8f acc;
#pragma unroll
  for (int q = 0; q < 8; ++q) acc[q] = 0.f;
  const int ksteps = Kpad >> 5;
  const half_t* srow = S + (long)(mt * 16) * Kpad;
  const half_t* xrow = XLT + (long)b * xbstride +
                       (long)(ft * 16 + (lane & 15)) * Kpad + ((lane < 16) ? 0 : 16);
  for (int ks = 0; ks < ksteps; ++ks) {
    v16h a = frag_a(srow + ks * 32, Kpad, lane, true);
    v16h bb;
#pragma unroll
    for (int i = 0; i < 16; ++i) bb[i] = xrow[ks * 32 + i];
    acc = wmma16(a, bb, acc);
  }
  const int col = ft * 16 + (lane & 15);
#pragma unroll
  for (int q = 0; q < 8; ++q) {
    int m = mt * 16 + q + ((lane < 16) ? 0 : 8);
    if (m < mvalid) {
      float v = acc[q];
      if (doRelu) v = fmaxf(v, 0.f);
      if (outMode == 0)
        outH[(long)b * obstride + (long)(row_off + m) * ld_out + col] = (half_t)v;
      else
        outF32[(long)b * obstride + (long)m * ld_out + col] = v;
    }
  }
}

// ---------------------------------------------------------------------------
// BN2 over batch + normalize to padded half [1024][1056]
// ---------------------------------------------------------------------------
__global__ void bn2_stats_kernel(const float* __restrict__ futf, const float* __restrict__ g,
                                 const float* __restrict__ bta, float* __restrict__ scale2,
                                 float* __restrict__ shift2) {
  int c = blockIdx.x, t = threadIdx.x;
  __shared__ float r1[256], r2[256];
  float s1 = 0.f, s2 = 0.f;
  for (int b = t; b < BB; b += 256) {
    float v = futf[(long)b * 1040 + c];
    s1 += v; s2 += v * v;
  }
  r1[t] = s1; r2[t] = s2; __syncthreads();
  for (int o = 128; o > 0; o >>= 1) {
    if (t < o) { r1[t] += r1[t + o]; r2[t] += r2[t + o]; }
    __syncthreads();
  }
  if (t == 0) {
    float mu = r1[0] / (float)BB;
    float var = r2[0] / (float)BB - mu * mu;
    float sc = g[c] * rsqrtf(var + BN_EPS);
    scale2[c] = sc;
    shift2[c] = bta[c] - mu * sc;
  }
}

__global__ void fnorm_kernel(const float* __restrict__ futf, const float* __restrict__ scale2,
                             const float* __restrict__ shift2, half_t* __restrict__ fn) {
  long i = (long)blockIdx.x * 256 + threadIdx.x;
  if (i >= (long)BB * 1056) return;
  int b = (int)(i / 1056), c = (int)(i % 1056);
  half_t v = (half_t)0.f;
  if (c < 1040) v = (half_t)(futf[(long)b * 1040 + c] * scale2[c] + shift2[c]);
  fn[i] = v;
}

// ---------------------------------------------------------------------------
// Fused final MLP (1056->128->128->65) with WMMA, one wave per 16 rows.
// ---------------------------------------------------------------------------
__global__ void __launch_bounds__(32) final_mlp_kernel(
    const half_t* __restrict__ fn,
    const half_t* __restrict__ Wf1h, const float* __restrict__ bf1,
    const half_t* __restrict__ Wf2h, const float* __restrict__ bf2,
    const half_t* __restrict__ Wf3h, const float* __restrict__ bf3,
    float* __restrict__ fout) {
  __shared__ half_t h1[16][128];
  __shared__ half_t h2[16][128];
  const int lane = threadIdx.x;
  const int tile = blockIdx.x;
  const int colL = lane & 15;
  const int hi = (lane < 16) ? 0 : 8;
  // layer 1: K=1056 (1040 valid), 128 out, A-frag reused over 8 col tiles
  v8f acc1[8];
#pragma unroll
  for (int ct = 0; ct < 8; ++ct)
#pragma unroll
    for (int q = 0; q < 8; ++q) acc1[ct][q] = bf1[ct * 16 + colL];
  const half_t* arow = fn + (long)tile * 16 * 1056;
  for (int ks = 0; ks < 33; ++ks) {
    v16h a = frag_a(arow + ks * 32, 1056, lane, true);
#pragma unroll
    for (int ct = 0; ct < 8; ++ct) {
      v16h bb = frag_b(Wf1h, 1056, ct * 16, ks * 32, lane);
      acc1[ct] = wmma16(a, bb, acc1[ct]);
    }
  }
#pragma unroll
  for (int ct = 0; ct < 8; ++ct)
#pragma unroll
    for (int q = 0; q < 8; ++q) h1[q + hi][ct * 16 + colL] = (half_t)fmaxf(acc1[ct][q], 0.f);
  __syncthreads();
  // layer 2: K=128
#pragma unroll
  for (int ct = 0; ct < 8; ++ct) {
    int col = ct * 16 + colL;
    v8f acc;
#pragma unroll
    for (int q = 0; q < 8; ++q) acc[q] = bf2[col];
#pragma unroll
    for (int ks = 0; ks < 4; ++ks) {
      v16h a  = frag_a(&h1[0][0] + ks * 32, 128, lane, true);
      v16h bb = frag_b(Wf2h, 128, ct * 16, ks * 32, lane);
      acc = wmma16(a, bb, acc);
    }
#pragma unroll
    for (int q = 0; q < 8; ++q) h2[q + hi][col] = (half_t)fmaxf(acc[q], 0.f);
  }
  __syncthreads();
  // layer 3: K=128, 65 valid out (pad 80)
#pragma unroll
  for (int ct = 0; ct < 5; ++ct) {
    int col = ct * 16 + colL;
    float bias = (col < FUTN) ? bf3[col] : 0.f;
    v8f acc;
#pragma unroll
    for (int q = 0; q < 8; ++q) acc[q] = bias;
#pragma unroll
    for (int ks = 0; ks < 4; ++ks) {
      v16h a  = frag_a(&h2[0][0] + ks * 32, 128, lane, true);
      v16h bb = frag_b(Wf3h, 128, ct * 16, ks * 32, lane);
      acc = wmma16(a, bb, acc);
    }
    if (col < FUTN) {
#pragma unroll
      for (int q = 0; q < 8; ++q) {
        long row = (long)tile * 16 + q + hi;
        fout[row * FUTN + col] = acc[q];
      }
    }
  }
}

// ---------------------------------------------------------------------------
// xd = dfull[n] * xcat (half, padded [288][32]), sq = |xd|^2 per row (f32)
// ---------------------------------------------------------------------------
__global__ void xd_prep_kernel(const half_t* __restrict__ xcat, const float* __restrict__ dfull,
                               half_t* __restrict__ xd, float* __restrict__ sq) {
  long i = (long)blockIdx.x * 256 + threadIdx.x;
  if (i >= (long)BB * 288) return;
  int n = (int)(i % 288);
  const half_t* src = xcat + i * 32;
  half_t* dst = xd + i * 32;
  float s = 0.f;
  if (n < NN) {
    float dn = dfull[n];
#pragma unroll
    for (int f = 0; f < 16; ++f) {
      float v = dn * (float)src[f];
      dst[f] = (half_t)v;
      s += v * v;
    }
#pragma unroll
    for (int f = 16; f < 32; ++f) dst[f] = (half_t)0.f;
  } else {
#pragma unroll
    for (int f = 0; f < 32; ++f) dst[f] = (half_t)0.f;
  }
  sq[i] = s;
}

// ---------------------------------------------------------------------------
// Pairwise distance via WMMA Gram: dist[b][i][j]
// ---------------------------------------------------------------------------
__global__ void __launch_bounds__(256) dist_kernel(const half_t* __restrict__ xd,
                                                   const float* __restrict__ sq,
                                                   float* __restrict__ dist) {
  const int it = blockIdx.x % 17;
  const int b = blockIdx.x / 17;
  const int wid = threadIdx.x >> 5;
  const int lane = threadIdx.x & 31;
  const half_t* xb = xd + (long)b * 288 * 32;
  const float* sqb = sq + (long)b * 288;
  v16h a = frag_a(xb + (long)it * 16 * 32, 32, lane, true);
  const int i0 = it * 16;
  for (int jt = wid; jt < 17; jt += 8) {
    v16h bb = frag_b(xb, 32, jt * 16, 0, lane);
    v8f acc;
#pragma unroll
    for (int q = 0; q < 8; ++q) acc[q] = 0.f;
    acc = wmma16(a, bb, acc);
    const int j = jt * 16 + (lane & 15);
    if (j < NN) {
      float sj = sqb[j];
#pragma unroll
      for (int q = 0; q < 8; ++q) {
        int i = i0 + q + ((lane < 16) ? 0 : 8);
        if (i < NN) {
          float sd = sqb[i] + sj - 2.f * acc[q];
          sd = fmaxf(sd, 0.f);
          dist[((long)b * NN + i) * NN + j] = (sd > 0.f) ? sqrtf(sd) : 0.f;
        }
      }
    }
  }
}

// ---------------------------------------------------------------------------
// Host launcher
// ---------------------------------------------------------------------------
static inline size_t align256(size_t x) { return (x + 255) & ~(size_t)255; }

extern "C" void kernel_launch(void* const* d_in, const int* in_sizes, int n_in,
                              void* d_out, int out_size, void* d_ws, size_t ws_size,
                              hipStream_t stream) {
  const float* data = (const float*)d_in[0];
  const float* emb0 = (const float*)d_in[1];
  const float* emb1 = (const float*)d_in[2];
  const float* emb2 = (const float*)d_in[3];
  const float* bn1g = (const float*)d_in[4];
  const float* bn1b = (const float*)d_in[5];
  const float* W1 = (const float*)d_in[6];
  const float* b1 = (const float*)d_in[7];
  const float* W2 = (const float*)d_in[8];
  const float* b2 = (const float*)d_in[9];
  const float* W3 = (const float*)d_in[10];
  const float* b3 = (const float*)d_in[11];
  const float* M = (const float*)d_in[12];
  const float* Wg1a = (const float*)d_in[13];
  const float* Wg1b = (const float*)d_in[14];
  const float* Wg2a = (const float*)d_in[15];
  const float* Wg2b = (const float*)d_in[16];
  const float* Wp = (const float*)d_in[17];
  const float* bn2g = (const float*)d_in[18];
  const float* bn2b = (const float*)d_in[19];
  const float* Wf1 = (const float*)d_in[20];
  const float* bf1 = (const float*)d_in[21];
  const float* Wf2 = (const float*)d_in[22];
  const float* bf2 = (const float*)d_in[23];
  const float* Wf3 = (const float*)d_in[24];
  const float* bf3 = (const float*)d_in[25];

  float* out_f = (float*)d_out;                                  // (1024, 65)
  float* out_dist = out_f + (long)BB * FUTN;                     // (1024, 265, 265)
  float* out_A = out_dist + (long)BB * NN * NN;                  // (265, 265)

  // ---- workspace layout -------------------------------------------------
  char* ws = (char*)d_ws;
  size_t off = 0;
  auto alloc = [&](size_t bytes) { char* p = ws + off; off = align256(off + bytes); return p; };

  float* dfull = (float*)alloc(NN * 4);
  float* d11 = (float*)alloc(224 * 4);
  float* d22 = (float*)alloc(96 * 4);
  float* dp = (float*)alloc(96 * 4);
  float* scale = (float*)alloc(NN * 4);
  float* shiftv = (float*)alloc(NN * 4);
  float* rs = (float*)alloc(70 * 4);
  float* rs2 = (float*)alloc(70 * 4);
  float* scale2 = (float*)alloc(1040 * 4);
  float* shift2 = (float*)alloc(1040 * 4);

  half_t* S11h = (half_t*)alloc((size_t)224 * 224 * 2);
  half_t* S22h = (half_t*)alloc((size_t)96 * 96 * 2);
  half_t* Sph = (half_t*)alloc((size_t)96 * 288 * 2);

  half_t* W1h = (half_t*)alloc((size_t)128 * 64 * 2);
  half_t* W2h = (half_t*)alloc((size_t)128 * 128 * 2);
  half_t* W3h = (half_t*)alloc((size_t)64 * 128 * 2);
  half_t* G1ah = (half_t*)alloc((size_t)64 * 64 * 2);
  half_t* G1bh = (half_t*)alloc((size_t)16 * 64 * 2);
  half_t* G2ah = (half_t*)alloc((size_t)64 * 64 * 2);
  half_t* G2bh = (half_t*)alloc((size_t)16 * 64 * 2);
  half_t* Wph = (half_t*)alloc((size_t)16 * 32 * 2);
  half_t* Wf1h = (half_t*)alloc((size_t)128 * 1056 * 2);
  half_t* Wf2h = (half_t*)alloc((size_t)128 * 128 * 2);
  half_t* Wf3h = (half_t*)alloc((size_t)80 * 128 * 2);

  // big regions (R0/R1 are reused after their first tenants die)
  char* R0 = alloc((size_t)BB * NN * 64 * 2);            // x_h; later xcat + xlpt
  char* R1 = alloc((size_t)BB * 64 * 224 * 2);           // xl1t; later xd/sq/futf/fn
  half_t* g1 = (half_t*)alloc((size_t)BB * 224 * 64 * 2);
  half_t* xl1bt = (half_t*)alloc((size_t)BB * 16 * 224 * 2);
  half_t* xl2t = (half_t*)alloc((size_t)BB * 64 * 96 * 2);
  half_t* g2 = (half_t*)alloc((size_t)BB * 96 * 64 * 2);
  half_t* xl2bt = (half_t*)alloc((size_t)BB * 16 * 96 * 2);

  half_t* x_h = (half_t*)R0;
  half_t* xcat = (half_t*)R0;                                            // 1024*288*32
  half_t* xlpt = (half_t*)(R0 + align256((size_t)BB * 288 * 32 * 2));    // 1024*16*288
  half_t* xl1t = (half_t*)R1;
  size_t r1o = 0;
  half_t* xd = (half_t*)(R1 + r1o); r1o += align256((size_t)BB * 288 * 32 * 2);
  float* sqv = (float*)(R1 + r1o); r1o += align256((size_t)BB * 288 * 4);
  float* futf = (float*)(R1 + r1o); r1o += align256((size_t)BB * FUTN * 16 * 4);
  half_t* fn = (half_t*)(R1 + r1o);

  auto cvt = [&](const float* s, half_t* d, int rows, int cols, int ldk, int rowsPad) {
    int total = rowsPad * ldk;
    cvt_pad_kernel<<<(total + 255) / 256, 256, 0, stream>>>(s, d, rows, cols, ldk, rowsPad);
  };

  // ---- weight prep ------------------------------------------------------
  cvt(W1, W1h, 128, 51, 64, 128);
  cvt(W2, W2h, 128, 128, 128, 128);
  cvt(W3, W3h, 50, 128, 128, 64);
  cvt(Wg1a, G1ah, 64, 51, 64, 64);
  cvt(Wg1b, G1bh, 16, 64, 64, 16);
  cvt(Wg2a, G2ah, 64, 50, 64, 64);
  cvt(Wg2b, G2bh, 16, 64, 64, 16);
  cvt(Wp, Wph, 16, 16, 32, 16);
  cvt(Wf1, Wf1h, 128, 1040, 1056, 128);
  cvt(Wf2, Wf2h, 128, 128, 128, 128);
  cvt(Wf3, Wf3h, 65, 128, 128, 80);

  // ---- adjacency / degrees / S matrices ---------------------------------
  adj_kernel<<<NN, 256, 0, stream>>>(M, out_A, dfull);
  subdeg_kernel<<<330, 256, 0, stream>>>(out_A, d11, d22, dp);
  {
    int total = 224 * 224 + 96 * 96 + 96 * 288;
    build_s_kernel<<<(total + 255) / 256, 256, 0, stream>>>(out_A, d11, d22, dp, S11h, S22h, Sph);
  }

  // ---- BN1 stats + fused pre-MLP ----------------------------------------
  embsum_kernel<<<1, 128, 0, stream>>>(emb0, emb1, emb2, rs, rs2);
  nodestats_kernel<<<NN, 256, 0, stream>>>(data, rs, rs2, bn1g, bn1b, scale, shiftv);
  pre_mlp_kernel<<<(BB * NN) / 16, 32, 0, stream>>>(data, emb0, emb1, emb2, scale, shiftv,
                                                    W1h, b1, W2h, b2, W3h, b3, x_h);

  // ---- GCN branch 1 & 2 --------------------------------------------------
  xl_gemm_kernel<4><<<BB * 14, 32, 0, stream>>>(x_h, (long)NN * 64, 64, 0, PASTN, 224,
                                                G1ah, 64, xl1t, (long)64 * 224);
  xl_gemm_kernel<4><<<BB * 6, 32, 0, stream>>>(x_h, (long)NN * 64, 64, PASTN, FUTN, 96,
                                               G2ah, 64, xl2t, (long)64 * 96);
  // x_h now dead: zero xcat alias region
  zero_half_kernel<<<(int)(((long)BB * 288 * 32 + 255) / 256), 256, 0, stream>>>(
      xcat, (long)BB * 288 * 32);

  agg_gemm_kernel<<<BB * 14 * 4, 32, 0, stream>>>(S11h, 224, 224, xl1t, (long)64 * 224, 64, 1,
                                                  g1, nullptr, 0, 64, (long)224 * 64, 0, 224);
  agg_gemm_kernel<<<BB * 6 * 4, 32, 0, stream>>>(S22h, 96, 96, xl2t, (long)64 * 96, 64, 1,
                                                 g2, nullptr, 0, 64, (long)96 * 64, 0, 96);

  xl_gemm_kernel<1><<<BB * 14, 32, 0, stream>>>(g1, (long)224 * 64, 64, 0, 224, 224,
                                                G1bh, 64, xl1bt, (long)16 * 224);
  xl_gemm_kernel<1><<<BB * 6, 32, 0, stream>>>(g2, (long)96 * 64, 64, 0, 96, 96,
                                               G2bh, 64, xl2bt, (long)16 * 96);

  agg_gemm_kernel<<<BB * 14, 32, 0, stream>>>(S11h, 224, 224, xl1bt, (long)16 * 224, 16, 0,
                                              xcat, nullptr, 0, 32, (long)288 * 32, 0, PASTN);
  agg_gemm_kernel<<<BB * 6, 32, 0, stream>>>(S22h, 96, 96, xl2bt, (long)16 * 96, 16, 0,
                                             xcat, nullptr, 0, 32, (long)288 * 32, PASTN, FUTN);

  // ---- future projection + BN2 + final MLP -------------------------------
  xl_gemm_kernel<1><<<BB * 18, 32, 0, stream>>>(xcat, (long)288 * 32, 32, 0, 288, 288,
                                                Wph, 32, xlpt, (long)16 * 288);
  agg_gemm_kernel<<<BB * 6, 32, 0, stream>>>(Sph, 288, 96, xlpt, (long)16 * 288, 16, 0,
                                             nullptr, futf, 1, 16, (long)FUTN * 16, 0, FUTN);
  bn2_stats_kernel<<<1040, 256, 0, stream>>>(futf, bn2g, bn2b, scale2, shift2);
  fnorm_kernel<<<(int)(((long)BB * 1056 + 255) / 256), 256, 0, stream>>>(futf, scale2, shift2, fn);
  final_mlp_kernel<<<BB / 16, 32, 0, stream>>>(fn, Wf1h, bf1, Wf2h, bf2, Wf3h, bf3, out_f);

  // ---- pairwise distance --------------------------------------------------
  xd_prep_kernel<<<(int)(((long)BB * 288 + 255) / 256), 256, 0, stream>>>(xcat, dfull, xd, sqv);
  dist_kernel<<<BB * 17, 256, 0, stream>>>(xd, sqv, out_dist);

  (void)in_sizes; (void)n_in; (void)out_size; (void)ws_size;
}
